// MLPBlock_54150947668246
// MI455X (gfx1250) — matrix operand order repaired and checked by reading, not compiled, not run on hardware
//
#include <hip/hip_runtime.h>
#include <stdint.h>

// ---------------------------------------------------------------------------
// MI455X / gfx1250 MoE MLP block.
// bf16 WMMA (v_wmma_f32_16x16x32_bf16), wave32, 256-thread (8-wave) blocks.
// A tiles are staged global->LDS by the Tensor Data Mover (tensor_load_to_lds,
// TENSORcnt, triple-buffered, depth-2 pipeline with s_wait_tensorcnt <=1).
// B tiles are register-staged + transposed into LDS one step ahead.
// ---------------------------------------------------------------------------

typedef __attribute__((ext_vector_type(16))) __bf16 v16bf;
typedef __attribute__((ext_vector_type(8)))  float  v8f;
typedef __attribute__((ext_vector_type(4)))  unsigned int su4;
typedef __attribute__((ext_vector_type(8)))  unsigned int su8;

#define H_DIM 2880
#define I_DIM 2880
#define TWOI  5760
#define T_DIM 512
#define E_NUM 8
#define K_TOP 4
#define NSTEP (H_DIM / 32)   // 90 K-steps for every GEMM here

// LDS strides (in ushort): A stride 40 (16-dword rows + 4-dword TDM pad),
// B-transposed stride 36. Both even -> dword-aligned pair reads.
#define AS_STRIDE 40
#define BT_STRIDE 36

static __device__ __forceinline__ float bf2f(unsigned short u) {
  union { unsigned int i; float f; } x; x.i = ((unsigned int)u) << 16; return x.f;
}
static __device__ __forceinline__ unsigned short f2bf(float f) {
  union { float f; unsigned int i; } x; x.f = f;
  unsigned int i = x.i;
  unsigned int r = (i + 0x7FFFu + ((i >> 16) & 1u)) >> 16;  // RNE
  return (unsigned short)r;
}
static __device__ __forceinline__ float sigmoidf(float x) {
  return 1.0f / (1.0f + __expf(-x));
}

union Frag16 { v16bf v; unsigned int u[8]; };

// --- CDNA5 Tensor Data Mover: 2D tile (2-byte elems) global -> LDS ---------
// D# per CDNA5 ISA. Group0: count=1|lds_addr|global_addr|
// type=2. Group1: data_size=1(2B), pad 4 dwords every 16 dwords (row=64B ->
// LDS row stride 80B == AS_STRIDE), tile 32x128, dim0_stride = ld.
static __device__ __forceinline__ void tdm_load_A32x128(const void* gsrc, void* ldst,
                                                        unsigned ld_elems) {
  const unsigned long long ga = (unsigned long long)(uintptr_t)gsrc;
  su4 g0;
  g0[0] = 1u;                                    // count=1, user descriptor
  g0[1] = (unsigned)(uintptr_t)ldst;             // LDS byte address
  g0[2] = (unsigned)(ga & 0xFFFFFFFFu);          // global addr [31:0]
  g0[3] = (unsigned)((ga >> 32) & 0x01FFFFFFu) | (2u << 30);  // [56:32] | type=2
  su8 g1;
  g1[0] = (1u << 16)                             // data_size = 2 bytes
        | (1u << 20)                             // pad_enable
        | (3u << 22)                             // pad_interval: 16 dwords (64B)
        | (3u << 25);                            // pad_amount: 4 dwords (16B)
  const unsigned d0 = ld_elems;                  // generous OOB extents
  const unsigned d1 = 0x100000u;
  g1[1] = (d0 & 0xFFFFu) << 16;                  // tensor_dim0[15:0]
  g1[2] = (d0 >> 16) | ((d1 & 0xFFFFu) << 16);   // dim0[31:16] | dim1[15:0]
  g1[3] = (d1 >> 16) | (32u << 16);              // dim1[31:16] | tile_dim0=32
  g1[4] = 128u;                                  // tile_dim1=128, tile_dim2=0
  g1[5] = ld_elems;                              // dim0_stride[31:0]
  g1[6] = 0u;                                    // dim0_stride[47:32], dim1_stride lo
  g1[7] = 0u;                                    // dim1_stride hi
  asm volatile("tensor_load_to_lds %0, %1" :: "s"(g0), "s"(g1) : "memory");
}

// --- B staging: 32 K x 64 cols; regs -> TRANSPOSED LDS sB[col][k] ----------
static __device__ __forceinline__ uint4 loadB_regs(const unsigned short* B,
                                                   int k0, int n0, int ldb, int tid) {
  const int kr = tid >> 3, grp = tid & 7;
  return *(const uint4*)(B + (size_t)(k0 + kr) * ldb + n0 + grp * 8);
}
static __device__ __forceinline__ void storeB_T(unsigned short* sB, uint4 d, int tid) {
  const int kr = tid >> 3, grp = tid & 7;
  const unsigned int w4[4] = {d.x, d.y, d.z, d.w};
#pragma unroll
  for (int q = 0; q < 8; ++q)
    sB[(grp * 8 + q) * BT_STRIDE + kr] = (unsigned short)(w4[q >> 1] >> ((q & 1) * 16));
}
// A fragment (16x32 bf16): lanes 0-15 row M=lane, K pairs 0..7 & 16..23;
// lanes 16-31 same rows, K pairs 8..15 & 24..31 (ISA 7.12.2).
static __device__ __forceinline__ void load_fragA(Frag16& f, const unsigned short* sA,
                                                  int wave, int lane) {
  const int lm = lane & 15, kb = ((lane >> 4) & 1) * 8;
  const unsigned short* arow = &sA[(wave * 16 + lm) * AS_STRIDE];
#pragma unroll
  for (int v = 0; v < 4; ++v) {
    f.u[v]     = *(const unsigned int*)&arow[kb + 2 * v];
    f.u[v + 4] = *(const unsigned int*)&arow[16 + kb + 2 * v];
  }
}
// B fragment (32x16 bf16): lanes 0-15 col N=lane K=0..7 & 16..23; lanes 16-31 K=8..15 & 24..31.
static __device__ __forceinline__ void load_fragB(Frag16& f, const unsigned short* sB,
                                                  int col, int lane) {
  const int kb = ((lane >> 4) & 1) * 8;
  const unsigned short* bcol = &sB[col * BT_STRIDE];
#pragma unroll
  for (int v = 0; v < 4; ++v) {
    f.u[v]     = *(const unsigned int*)&bcol[kb + 2 * v];
    f.u[v + 4] = *(const unsigned int*)&bcol[16 + kb + 2 * v];
  }
}
static __device__ __forceinline__ v8f wmma_bf16(const Frag16& a, const Frag16& b, v8f c) {
  return __builtin_amdgcn_wmma_f32_16x16x32_bf16(false, a.v, false, b.v,
                                                 (short)0, c, false, false);
}
// Wait so that the OLDER of (up to) two in-flight TDM loads is complete.
static __device__ __forceinline__ void tdm_wait(bool two_in_flight) {
  if (two_in_flight) __builtin_amdgcn_s_wait_tensorcnt(1);
  else               __builtin_amdgcn_s_wait_tensorcnt(0);
}

// ---------------------------------------------------------------------------
// Kernel 1: xc = x + context @ Wc + bc      [512,2880] f32 out
// ---------------------------------------------------------------------------
__global__ __launch_bounds__(256) void k_ctx_gemm(const unsigned short* __restrict__ x,
                                                  const unsigned short* __restrict__ ctx,
                                                  const unsigned short* __restrict__ Wc,
                                                  const unsigned short* __restrict__ bc,
                                                  float* __restrict__ xc) {
  __shared__ unsigned short sA[3][128 * AS_STRIDE];
  __shared__ unsigned short sB[2][64 * BT_STRIDE];
  const int tid = threadIdx.x, wave = tid >> 5, lane = tid & 31;
  const int lm = lane & 15, rsel = (lane >> 4) & 1;
  const int m0 = blockIdx.y * 128, n0 = blockIdx.x * 64;
  const unsigned short* Arow = ctx + (size_t)m0 * H_DIM;

  v8f acc[4];
#pragma unroll
  for (int j = 0; j < 4; ++j) acc[j] = (v8f)0.0f;

  // prologue: TDM steps 0,1 into sA[0],sA[1]; B step 0 into sB[0]
  if (wave == 0) {
    tdm_load_A32x128(Arow + 0,  sA[0], H_DIM);
    tdm_load_A32x128(Arow + 32, sA[1], H_DIM);
  }
  storeB_T(sB[0], loadB_regs(Wc, 0, n0, H_DIM, tid), tid);
  if (wave == 0) tdm_wait(true);
  __syncthreads();

  for (int i = 0; i < NSTEP; ++i) {
    const int k0 = i * 32;
    const int bufA = i % 3, bufB = i & 1;
    if (wave == 0 && i + 2 < NSTEP)
      tdm_load_A32x128(Arow + (k0 + 64), sA[(i + 2) % 3], H_DIM);
    uint4 bnx;
    if (i + 1 < NSTEP) {
      bnx = loadB_regs(Wc, k0 + 32, n0, H_DIM, tid);
      if (k0 + 64 < H_DIM)
        __builtin_prefetch(Wc + (size_t)(k0 + 64 + (tid >> 3)) * H_DIM + n0 + (tid & 7) * 8, 0, 1);
    }
    // preload all fragments, then run WMMAs back-to-back
    Frag16 afr; load_fragA(afr, sA[bufA], wave, lane);
    Frag16 bfr[4];
#pragma unroll
    for (int j = 0; j < 4; ++j) load_fragB(bfr[j], sB[bufB], j * 16 + lm, lane);
#pragma unroll
    for (int j = 0; j < 4; ++j) acc[j] = wmma_bf16(afr, bfr[j], acc[j]);
    if (i + 1 < NSTEP) storeB_T(sB[(i + 1) & 1], bnx, tid);
    if (wave == 0) tdm_wait(i + 2 < NSTEP);
    __syncthreads();
  }
#pragma unroll
  for (int j = 0; j < 4; ++j) {
    const int col = n0 + j * 16 + lm;
    const float bias = bf2f(bc[col]);
#pragma unroll
    for (int r = 0; r < 8; ++r) {
      const int row = m0 + wave * 16 + r + rsel * 8;
      xc[(size_t)row * H_DIM + col] = acc[j][r] + bias + bf2f(x[(size_t)row * H_DIM + col]);
    }
  }
}

// ---------------------------------------------------------------------------
// Kernel 2: t = rmsnorm(xc) * scale  -> bf16
// ---------------------------------------------------------------------------
__global__ __launch_bounds__(256) void k_rmsnorm(const float* __restrict__ xc,
                                                 const float* __restrict__ scale,
                                                 unsigned short* __restrict__ t) {
  __shared__ float red[256];
  const int row = blockIdx.x, tid = threadIdx.x;
  const float* xr = xc + (size_t)row * H_DIM;
  float s = 0.f;
  for (int h = tid; h < H_DIM; h += 256) { float v = xr[h]; s += v * v; }
  red[tid] = s; __syncthreads();
  for (int off = 128; off > 0; off >>= 1) {
    if (tid < off) red[tid] += red[tid + off];
    __syncthreads();
  }
  const float inv = rsqrtf(red[0] / (float)H_DIM + 1e-5f);
  for (int h = tid; h < H_DIM; h += 256)
    t[(size_t)row * H_DIM + h] = f2bf(xr[h] * inv * scale[h]);
}

// ---------------------------------------------------------------------------
// Kernel 3: router: g = t@Wg + bg, top-4 softmax -> dense cw[T,8]
// ---------------------------------------------------------------------------
__global__ __launch_bounds__(256) void k_route(const unsigned short* __restrict__ t,
                                               const unsigned short* __restrict__ Wg,
                                               const unsigned short* __restrict__ bg,
                                               float* __restrict__ cw) {
  __shared__ float sacc[256][E_NUM];
  __shared__ float g[E_NUM];
  const int row = blockIdx.x, tid = threadIdx.x;
  float acc[E_NUM];
#pragma unroll
  for (int e = 0; e < E_NUM; ++e) acc[e] = 0.f;
  for (int h = tid; h < H_DIM; h += 256) {
    const float tv = bf2f(t[(size_t)row * H_DIM + h]);
    const uint4 wr = *(const uint4*)(Wg + (size_t)h * E_NUM);
    const unsigned int w4[4] = {wr.x, wr.y, wr.z, wr.w};
#pragma unroll
    for (int q = 0; q < 8; ++q)
      acc[q] += tv * bf2f((unsigned short)(w4[q >> 1] >> ((q & 1) * 16)));
  }
#pragma unroll
  for (int e = 0; e < E_NUM; ++e) sacc[tid][e] = acc[e];
  __syncthreads();
  if (tid < E_NUM) {
    float s = bf2f(bg[tid]);
    for (int i = 0; i < 256; ++i) s += sacc[i][tid];
    g[tid] = s;
  }
  __syncthreads();
  if (tid == 0) {
    float vals[K_TOP]; int idx[K_TOP]; unsigned taken = 0;
    for (int k = 0; k < K_TOP; ++k) {
      float best = -3.4e38f; int bi = 0;
      for (int e = 0; e < E_NUM; ++e)
        if (!((taken >> e) & 1u) && g[e] > best) { best = g[e]; bi = e; }
      vals[k] = best; idx[k] = bi; taken |= 1u << bi;
    }
    float ex[K_TOP], s = 0.f;
    for (int k = 0; k < K_TOP; ++k) { ex[k] = __expf(vals[k] - vals[0]); s += ex[k]; }
    float* o = cw + (size_t)row * E_NUM;
    for (int e = 0; e < E_NUM; ++e) o[e] = 0.f;
    for (int k = 0; k < K_TOP; ++k) o[idx[k]] = ex[k] / s;
  }
}

// ---------------------------------------------------------------------------
// Kernel 4: per-expert up-proj + fused interleaved SwiGLU
//   a[e] = swiglu(t @ W1[e] + b1[e])  -> bf16 [E,T,I]
// Block covers 128 rows x 64 W1 columns (= 32 glu + 32 lin interleaved).
// ---------------------------------------------------------------------------
__global__ __launch_bounds__(256) void k_up_swiglu(const unsigned short* __restrict__ t,
                                                   const unsigned short* __restrict__ W1,
                                                   const unsigned short* __restrict__ b1,
                                                   unsigned short* __restrict__ aws) {
  __shared__ unsigned short sA[3][128 * AS_STRIDE];
  __shared__ unsigned short sB[2][64 * BT_STRIDE];
  const int tid = threadIdx.x, wave = tid >> 5, lane = tid & 31;
  const int lm = lane & 15, rsel = (lane >> 4) & 1;
  const int e = blockIdx.z;
  const int m0 = blockIdx.y * 128;
  const int n0h = blockIdx.x * 64;   // W1 column base (interleaved space)
  const int n0a = blockIdx.x * 32;   // a column base
  const unsigned short* W1e = W1 + (size_t)e * H_DIM * TWOI;
  const unsigned short* Arow = t + (size_t)m0 * H_DIM;

  v8f accg[2], accl[2];
#pragma unroll
  for (int j = 0; j < 2; ++j) { accg[j] = (v8f)0.0f; accl[j] = (v8f)0.0f; }

  if (wave == 0) {
    tdm_load_A32x128(Arow + 0,  sA[0], H_DIM);
    tdm_load_A32x128(Arow + 32, sA[1], H_DIM);
  }
  storeB_T(sB[0], loadB_regs(W1e, 0, n0h, TWOI, tid), tid);
  if (wave == 0) tdm_wait(true);
  __syncthreads();

  for (int i = 0; i < NSTEP; ++i) {
    const int k0 = i * 32;
    const int bufA = i % 3, bufB = i & 1;
    if (wave == 0 && i + 2 < NSTEP)
      tdm_load_A32x128(Arow + (k0 + 64), sA[(i + 2) % 3], H_DIM);
    uint4 bnx;
    if (i + 1 < NSTEP) {
      bnx = loadB_regs(W1e, k0 + 32, n0h, TWOI, tid);
      if (k0 + 64 < H_DIM)
        __builtin_prefetch(W1e + (size_t)(k0 + 64 + (tid >> 3)) * TWOI + n0h + (tid & 7) * 8, 0, 1);
    }
    Frag16 afr; load_fragA(afr, sA[bufA], wave, lane);
    Frag16 bfr[4];
#pragma unroll
    for (int j = 0; j < 2; ++j) {
      load_fragB(bfr[2 * j + 0], sB[bufB], j * 32 + 2 * lm + 0, lane);  // glu
      load_fragB(bfr[2 * j + 1], sB[bufB], j * 32 + 2 * lm + 1, lane);  // lin
    }
#pragma unroll
    for (int j = 0; j < 2; ++j) {
      accg[j] = wmma_bf16(afr, bfr[2 * j + 0], accg[j]);
      accl[j] = wmma_bf16(afr, bfr[2 * j + 1], accl[j]);
    }
    if (i + 1 < NSTEP) storeB_T(sB[(i + 1) & 1], bnx, tid);
    if (wave == 0) tdm_wait(i + 2 < NSTEP);
    __syncthreads();
  }
#pragma unroll
  for (int j = 0; j < 2; ++j) {
    const int ng = n0a + j * 16 + lm;                 // a-column
    const float biasg = bf2f(b1[(size_t)e * TWOI + 2 * ng]);
    const float biasl = bf2f(b1[(size_t)e * TWOI + 2 * ng + 1]);
#pragma unroll
    for (int r = 0; r < 8; ++r) {
      const int row = m0 + wave * 16 + r + rsel * 8;
      float gv = accg[j][r] + biasg;
      float lv = accl[j][r] + biasl;
      gv = fminf(gv, 7.0f);
      lv = fminf(fmaxf(lv, -7.0f), 7.0f);
      const float av = gv * sigmoidf(1.702f * gv) * (lv + 1.0f);
      aws[((size_t)e * T_DIM + row) * I_DIM + ng] = f2bf(av);
    }
  }
}

// ---------------------------------------------------------------------------
// Kernel 5: out = x + sum_e cw[t,e] * (a[e] @ W2[e] + b2[e])   -> bf16
// ---------------------------------------------------------------------------
__global__ __launch_bounds__(256) void k_down_combine(const unsigned short* __restrict__ aws,
                                                      const unsigned short* __restrict__ W2,
                                                      const unsigned short* __restrict__ b2,
                                                      const float* __restrict__ cw,
                                                      const unsigned short* __restrict__ x,
                                                      unsigned short* __restrict__ out) {
  __shared__ unsigned short sA[3][128 * AS_STRIDE];
  __shared__ unsigned short sB[2][64 * BT_STRIDE];
  __shared__ float sCW[128 * E_NUM];
  const int tid = threadIdx.x, wave = tid >> 5, lane = tid & 31;
  const int lm = lane & 15, rsel = (lane >> 4) & 1;
  const int m0 = blockIdx.y * 128, n0 = blockIdx.x * 64;

  for (int s = tid; s < 128 * E_NUM; s += 256)
    sCW[s] = cw[(size_t)(m0 + (s >> 3)) * E_NUM + (s & 7)];

  v8f oacc[4];
#pragma unroll
  for (int j = 0; j < 4; ++j) oacc[j] = (v8f)0.0f;

  for (int e = 0; e < E_NUM; ++e) {
    const unsigned short* Ae = aws + (size_t)e * T_DIM * I_DIM;
    const unsigned short* Be = W2 + (size_t)e * I_DIM * H_DIM;
    const unsigned short* Arow = Ae + (size_t)m0 * I_DIM;
    v8f acc[4];
#pragma unroll
    for (int j = 0; j < 4; ++j) acc[j] = (v8f)0.0f;

    if (wave == 0) {
      tdm_load_A32x128(Arow + 0,  sA[0], I_DIM);
      tdm_load_A32x128(Arow + 32, sA[1], I_DIM);
    }
    storeB_T(sB[0], loadB_regs(Be, 0, n0, H_DIM, tid), tid);
    if (wave == 0) tdm_wait(true);
    __syncthreads();

    for (int i = 0; i < NSTEP; ++i) {
      const int k0 = i * 32;
      const int bufA = i % 3, bufB = i & 1;
      if (wave == 0 && i + 2 < NSTEP)
        tdm_load_A32x128(Arow + (k0 + 64), sA[(i + 2) % 3], I_DIM);
      uint4 bnx;
      if (i + 1 < NSTEP) {
        bnx = loadB_regs(Be, k0 + 32, n0, H_DIM, tid);
        if (k0 + 64 < I_DIM)
          __builtin_prefetch(Be + (size_t)(k0 + 64 + (tid >> 3)) * H_DIM + n0 + (tid & 7) * 8, 0, 1);
      }
      Frag16 afr; load_fragA(afr, sA[bufA], wave, lane);
      Frag16 bfr[4];
#pragma unroll
      for (int j = 0; j < 4; ++j) load_fragB(bfr[j], sB[bufB], j * 16 + lm, lane);
#pragma unroll
      for (int j = 0; j < 4; ++j) acc[j] = wmma_bf16(afr, bfr[j], acc[j]);
      if (i + 1 < NSTEP) storeB_T(sB[(i + 1) & 1], bnx, tid);
      if (wave == 0) tdm_wait(i + 2 < NSTEP);
      __syncthreads();
    }
#pragma unroll
    for (int j = 0; j < 4; ++j) {
      const int col = n0 + j * 16 + lm;
      const float bias = bf2f(b2[(size_t)e * H_DIM + col]);
#pragma unroll
      for (int r = 0; r < 8; ++r) {
        const int rl = wave * 16 + r + rsel * 8;
        const float we = sCW[rl * E_NUM + e];
        oacc[j][r] += we * (acc[j][r] + bias);
      }
    }
  }
#pragma unroll
  for (int j = 0; j < 4; ++j) {
    const int col = n0 + j * 16 + lm;
#pragma unroll
    for (int r = 0; r < 8; ++r) {
      const int row = m0 + wave * 16 + r + rsel * 8;
      out[(size_t)row * H_DIM + col] = f2bf(bf2f(x[(size_t)row * H_DIM + col]) + oacc[j][r]);
    }
  }
}

// ---------------------------------------------------------------------------
extern "C" void kernel_launch(void* const* d_in, const int* in_sizes, int n_in,
                              void* d_out, int out_size, void* d_ws, size_t ws_size,
                              hipStream_t stream) {
  (void)in_sizes; (void)n_in; (void)out_size; (void)ws_size;
  const unsigned short* x     = (const unsigned short*)d_in[0];
  const unsigned short* ctx   = (const unsigned short*)d_in[1];
  const unsigned short* Wc    = (const unsigned short*)d_in[2];
  const unsigned short* bc    = (const unsigned short*)d_in[3];
  const float*          scale = (const float*)d_in[4];
  const unsigned short* Wg    = (const unsigned short*)d_in[5];
  const unsigned short* bg    = (const unsigned short*)d_in[6];
  const unsigned short* W1    = (const unsigned short*)d_in[7];
  const unsigned short* b1    = (const unsigned short*)d_in[8];
  const unsigned short* W2    = (const unsigned short*)d_in[9];
  const unsigned short* b2    = (const unsigned short*)d_in[10];
  unsigned short* out = (unsigned short*)d_out;

  // workspace layout (all offsets 256B aligned)
  char* ws = (char*)d_ws;
  float*          xc  = (float*)ws;                                        // 512*2880*4  = 5,898,240
  unsigned short* t   = (unsigned short*)(ws + 5898240);                   // 512*2880*2  = 2,949,120
  float*          cwp = (float*)(ws + 5898240 + 2949120);                  // 512*8*4     = 16,384
  unsigned short* aws = (unsigned short*)(ws + 5898240 + 2949120 + 16384); // 8*512*2880*2 = 23,592,960

  k_ctx_gemm   <<<dim3(H_DIM / 64, T_DIM / 128), 256, 0, stream>>>(x, ctx, Wc, bc, xc);
  k_rmsnorm    <<<dim3(T_DIM),                   256, 0, stream>>>(xc, scale, t);
  k_route      <<<dim3(T_DIM),                   256, 0, stream>>>(t, Wg, bg, cwp);
  k_up_swiglu  <<<dim3(I_DIM / 32, T_DIM / 128, E_NUM), 256, 0, stream>>>(t, W1, b1, aws);
  k_down_combine<<<dim3(H_DIM / 64, T_DIM / 128), 256, 0, stream>>>(aws, W2, b2, cwp, x, out);
}
